// c_transform_52578989637676
// MI455X (gfx1250) — compile-verified, matrix operand order checked by reading, not run on hardware
//
#include <hip/hip_runtime.h>
#include <hip/hip_bf16.h>

// ---------------------------------------------------------------------------
// Problem constants (from setup_inputs: x[8192,512], target[8192,512], W[512,128])
// ---------------------------------------------------------------------------
#define N_PTS   8192
#define M_PTS   8192
#define D_DIM   512
#define K_DIM   128
#define KK_STEPS (D_DIM / 32)          // 16 wmma steps of K=32 over the D contraction
#define WAVES_PER_BLOCK 4              // 4 waves -> 64 target-columns per block
#define NSPLIT  8                      // split the N (min-reduction) dimension 8 ways

typedef __bf16 bf16_t;
typedef __attribute__((ext_vector_type(16))) __bf16 v16bf;
typedef __attribute__((ext_vector_type(8)))  __bf16 v8bf;
typedef __attribute__((ext_vector_type(8)))  float  v8f;
typedef __attribute__((ext_vector_type(4)))  float  v4f;

// ---------------------------------------------------------------------------
// bf16 helpers (bit-level, round-to-nearest-even; avoids __bf16 arithmetic)
// ---------------------------------------------------------------------------
__device__ __forceinline__ bf16_t f2bf(float f) {
    unsigned u = __float_as_uint(f);
    unsigned r = (u + 0x7FFFu + ((u >> 16) & 1u)) >> 16;
    unsigned short s = (unsigned short)r;
    return __builtin_bit_cast(bf16_t, s);
}
__device__ __forceinline__ float bf2f(bf16_t b) {
    unsigned short s = __builtin_bit_cast(unsigned short, b);
    return __uint_as_float(((unsigned)s) << 16);
}

// Build a 16-element A/B fragment from two contiguous 16B chunks at p and p+16 elems.
// (Per-lane WMMA 16-bit fragment layout: elems 0..7 = K[k0..k0+7], 8..15 = K[k0+16..k0+23])
__device__ __forceinline__ v16bf load_frag(const bf16_t* p) {
    v8bf lo = *(const v8bf*)(p);
    v8bf hi = *(const v8bf*)(p + 16);
    v16bf r;
#pragma unroll
    for (int i = 0; i < 8; ++i) { r[i] = lo[i]; r[i + 8] = hi[i]; }
    return r;
}

#define WMMA_BF16(A, B, C) \
    __builtin_amdgcn_wmma_f32_16x16x32_bf16(false, (A), false, (B), (short)0, (C), false, false)

// Per-16x16-tile epilogue: min over the 8 row-entries this lane holds of phi_b[n]-S.
__device__ __forceinline__ float tile_min(const v8f acc, const float* pb) {
    v4f p0 = *(const v4f*)(pb);
    v4f p1 = *(const v4f*)(pb + 4);
    float m = fminf(fminf(p0[0] - acc[0], p0[1] - acc[1]),
                    fminf(p0[2] - acc[2], p0[3] - acc[3]));
    m = fminf(m, fminf(fminf(p1[0] - acc[4], p1[1] - acc[5]),
                       fminf(p1[2] - acc[6], p1[3] - acc[7])));
    return m;
}

// ---------------------------------------------------------------------------
// Kernel 1: split fp32 array into bf16 hi/lo pair (x = hi + lo, |lo| ~ 2^-8 |x|)
// ---------------------------------------------------------------------------
__global__ void split_bf16_kernel(const float* __restrict__ src,
                                  bf16_t* __restrict__ hi, bf16_t* __restrict__ lo,
                                  long n) {
    long i = (long)blockIdx.x * blockDim.x + threadIdx.x;
    long stride = (long)gridDim.x * blockDim.x;
    for (; i < n; i += stride) {
        float v = src[i];
        bf16_t h = f2bf(v);
        hi[i] = h;
        lo[i] = f2bf(v - bf2f(h));
    }
}

// ---------------------------------------------------------------------------
// Kernel 2: per-row squared norm of target -> t_sq[M]
// ---------------------------------------------------------------------------
__global__ void rowsq_kernel(const float* __restrict__ t, float* __restrict__ sq) {
    __shared__ float red[256];
    const float* row = t + (size_t)blockIdx.x * D_DIM;
    float s = 0.f;
    for (int d = threadIdx.x; d < D_DIM; d += 256) { float v = row[d]; s += v * v; }
    red[threadIdx.x] = s;
    __syncthreads();
    for (int w = 128; w > 0; w >>= 1) {
        if (threadIdx.x < w) red[threadIdx.x] += red[threadIdx.x + w];
        __syncthreads();
    }
    if (threadIdx.x == 0) sq[blockIdx.x] = red[0];
}

// ---------------------------------------------------------------------------
// Kernel 3: per-row  phi_b = logsumexp(x @ W),  phi_k = 0.5||x||^2 - phi_b
// One 128-thread block per row; thread k owns logit k. 0.54 GFLOP total -> VALU ok.
// ---------------------------------------------------------------------------
__global__ void phi_kernel(const float* __restrict__ x, const float* __restrict__ W,
                           float* __restrict__ phi_b, float* __restrict__ phi_k) {
    __shared__ float xs[D_DIM];
    __shared__ float red[K_DIM];
    const int tid = threadIdx.x;
    const float* row = x + (size_t)blockIdx.x * D_DIM;

    float sq = 0.f;
    for (int d = tid; d < D_DIM; d += K_DIM) { float v = row[d]; xs[d] = v; sq += v * v; }
    red[tid] = sq;
    __syncthreads();
    for (int w = K_DIM / 2; w > 0; w >>= 1) {
        if (tid < w) red[tid] += red[tid + w];
        __syncthreads();
    }
    float xsq = red[0];
    __syncthreads();

    float acc = 0.f;
#pragma unroll 8
    for (int d = 0; d < D_DIM; ++d) acc = fmaf(xs[d], W[(size_t)d * K_DIM + tid], acc);

    // logsumexp over the 128 logits
    red[tid] = acc;
    __syncthreads();
    for (int w = K_DIM / 2; w > 0; w >>= 1) {
        if (tid < w) red[tid] = fmaxf(red[tid], red[tid + w]);
        __syncthreads();
    }
    float mx = red[0];
    __syncthreads();
    red[tid] = __expf(acc - mx);
    __syncthreads();
    for (int w = K_DIM / 2; w > 0; w >>= 1) {
        if (tid < w) red[tid] += red[tid + w];
        __syncthreads();
    }
    if (tid == 0) {
        float lse = mx + __logf(red[0]);
        phi_b[blockIdx.x] = lse;
        phi_k[blockIdx.x] = 0.5f * xsq - lse;
    }
}

// ---------------------------------------------------------------------------
// Kernel 4 (main): fused  S = x @ target^T  (split-bf16, 3 WMMA terms) with
// column-min epilogue:  colmin_part[split][m] = min_{n in chunk} (phi_b[n] - S[n,m])
//
// Each wave owns a 16-column strip of target; the full B tile (16 cols x 512 D,
// hi+lo) lives in ~256 VGPRs for the whole kernel. The n-loop processes TWO
// 16-row x-tiles per iteration (two independent WMMA accumulator chains) and
// software-pipelines the A-fragment loads one k-step ahead.
// ---------------------------------------------------------------------------
__global__ void __launch_bounds__(WAVES_PER_BLOCK * 32)
ctransform_min_kernel(const bf16_t* __restrict__ xh, const bf16_t* __restrict__ xl,
                      const bf16_t* __restrict__ th, const bf16_t* __restrict__ tl,
                      const float* __restrict__ phi_b,
                      float* __restrict__ colmin_part) {
    const int lane = threadIdx.x & 31;
    const int wave = threadIdx.x >> 5;
    const int col  = lane & 15;     // column / row within 16-tile
    const int hif  = lane >> 4;     // which k-half this lane holds
    const int mtile = blockIdx.x * (WAVES_PER_BLOCK * 16) + wave * 16;

    // ---- Load the full B tile (target hi/lo fragments) into registers.
    v16bf bh[KK_STEPS], bl[KK_STEPS];
    {
        const size_t rrow = (size_t)(mtile + col) * D_DIM + hif * 8;
#pragma unroll
        for (int kk = 0; kk < KK_STEPS; ++kk) {
            bh[kk] = load_frag(th + rrow + kk * 32);
            bl[kk] = load_frag(tl + rrow + kk * 32);
        }
    }

    const int n_chunk = N_PTS / NSPLIT;     // 1024, multiple of 32
    const int n_begin = blockIdx.y * n_chunk;
    const int n_end   = n_begin + n_chunk;

    float colmin = INFINITY;

    for (int n0 = n_begin; n0 < n_end; n0 += 32) {
        const size_t base0 = (size_t)(n0 + col) * D_DIM + hif * 8;
        const bf16_t* pxh0 = xh + base0;
        const bf16_t* pxl0 = xl + base0;
        const bf16_t* pxh1 = pxh0 + (size_t)16 * D_DIM;
        const bf16_t* pxl1 = pxl0 + (size_t)16 * D_DIM;

        // prefetch the next 32-row A block while computing this one
        if (n0 + 32 < n_end) {
            __builtin_prefetch(pxh0 + (size_t)32 * D_DIM, 0, 0);
            __builtin_prefetch(pxl0 + (size_t)32 * D_DIM, 0, 0);
        }

        v8f acc0 = {};
        v8f acc1 = {};

        // prologue: fragments for kk = 0
        v16bf ah0 = load_frag(pxh0);
        v16bf al0 = load_frag(pxl0);
        v16bf ah1 = load_frag(pxh1);
        v16bf al1 = load_frag(pxl1);

#pragma unroll
        for (int kk = 0; kk < KK_STEPS; ++kk) {
            // issue next k-step's loads before consuming the current fragments
            v16bf nh0, nl0, nh1, nl1;
            if (kk + 1 < KK_STEPS) {
                nh0 = load_frag(pxh0 + (kk + 1) * 32);
                nl0 = load_frag(pxl0 + (kk + 1) * 32);
                nh1 = load_frag(pxh1 + (kk + 1) * 32);
                nl1 = load_frag(pxl1 + (kk + 1) * 32);
            }
            // S += xh*th + xh*tl + xl*th   (xl*tl term ~2^-16, dropped);
            // interleave the two independent accumulator chains.
            acc0 = WMMA_BF16(ah0, bh[kk], acc0);
            acc1 = WMMA_BF16(ah1, bh[kk], acc1);
            acc0 = WMMA_BF16(ah0, bl[kk], acc0);
            acc1 = WMMA_BF16(ah1, bl[kk], acc1);
            acc0 = WMMA_BF16(al0, bh[kk], acc0);
            acc1 = WMMA_BF16(al1, bh[kk], acc1);
            ah0 = nh0; al0 = nl0; ah1 = nh1; al1 = nl1;
        }

        // Epilogue: acc[i] = S[n_tile + 8*hif + i][mtile + col]
        float m0 = tile_min(acc0, phi_b + n0 + 8 * hif);
        float m1 = tile_min(acc1, phi_b + n0 + 16 + 8 * hif);
        float m = fminf(m0, m1);
        m = fminf(m, __shfl_xor(m, 16, 32));   // merge the two row-halves of this column
        colmin = fminf(colmin, m);
    }

    if (hif == 0)
        colmin_part[(size_t)blockIdx.y * M_PTS + mtile + col] = colmin;
}

// ---------------------------------------------------------------------------
// Kernel 5: finalize -> out = mean(phi_k) + mean(0.5*t_sq + min over splits)
// ---------------------------------------------------------------------------
__global__ void finalize_kernel(const float* __restrict__ phi_k,
                                const float* __restrict__ t_sq,
                                const float* __restrict__ colmin_part,
                                float* __restrict__ out) {
    __shared__ float rk[256];
    __shared__ float rc[256];
    const int tid = threadIdx.x;

    float sk = 0.f;
    for (int i = tid; i < N_PTS; i += 256) sk += phi_k[i];

    float sc = 0.f;
    for (int m = tid; m < M_PTS; m += 256) {
        float mn = INFINITY;
#pragma unroll
        for (int s = 0; s < NSPLIT; ++s)
            mn = fminf(mn, colmin_part[(size_t)s * M_PTS + m]);
        sc += 0.5f * t_sq[m] + mn;
    }
    rk[tid] = sk; rc[tid] = sc;
    __syncthreads();
    for (int w = 128; w > 0; w >>= 1) {
        if (tid < w) { rk[tid] += rk[tid + w]; rc[tid] += rc[tid + w]; }
        __syncthreads();
    }
    if (tid == 0)
        out[0] = rk[0] / (float)N_PTS + rc[0] / (float)M_PTS;
}

// ---------------------------------------------------------------------------
// Host launcher
// ---------------------------------------------------------------------------
extern "C" void kernel_launch(void* const* d_in, const int* in_sizes, int n_in,
                              void* d_out, int out_size, void* d_ws, size_t ws_size,
                              hipStream_t stream) {
    const float* x      = (const float*)d_in[0];   // [N, D] fp32
    const float* target = (const float*)d_in[1];   // [M, D] fp32
    const float* W      = (const float*)d_in[2];   // [D, K] fp32
    float* out = (float*)d_out;

    // Workspace carve-up (~34 MB)
    char* p = (char*)d_ws;
    bf16_t* xh = (bf16_t*)p;  p += (size_t)N_PTS * D_DIM * sizeof(bf16_t);
    bf16_t* xl = (bf16_t*)p;  p += (size_t)N_PTS * D_DIM * sizeof(bf16_t);
    bf16_t* th = (bf16_t*)p;  p += (size_t)M_PTS * D_DIM * sizeof(bf16_t);
    bf16_t* tl = (bf16_t*)p;  p += (size_t)M_PTS * D_DIM * sizeof(bf16_t);
    float* phi_b = (float*)p; p += (size_t)N_PTS * sizeof(float);
    float* phi_k = (float*)p; p += (size_t)N_PTS * sizeof(float);
    float* t_sq  = (float*)p; p += (size_t)M_PTS * sizeof(float);
    float* colmin_part = (float*)p; p += (size_t)NSPLIT * M_PTS * sizeof(float);

    split_bf16_kernel<<<2048, 256, 0, stream>>>(x, xh, xl, (long)N_PTS * D_DIM);
    split_bf16_kernel<<<2048, 256, 0, stream>>>(target, th, tl, (long)M_PTS * D_DIM);
    rowsq_kernel<<<M_PTS, 256, 0, stream>>>(target, t_sq);
    phi_kernel<<<N_PTS, K_DIM, 0, stream>>>(x, W, phi_b, phi_k);

    dim3 grid(M_PTS / (WAVES_PER_BLOCK * 16), NSPLIT);
    ctransform_min_kernel<<<grid, WAVES_PER_BLOCK * 32, 0, stream>>>(
        xh, xl, th, tl, phi_b, colmin_part);

    finalize_kernel<<<1, 256, 0, stream>>>(phi_k, t_sq, colmin_part, out);
}
